// SiglipAttention_1511828488756
// MI455X (gfx1250) — matrix operand order repaired and checked by reading, not compiled, not run on hardware
//
#include <hip/hip_runtime.h>
#include <hip/hip_bf16.h>

typedef unsigned short u16;
typedef __attribute__((ext_vector_type(16))) __bf16 bf16x16;
typedef __attribute__((ext_vector_type(8)))  float   f32x8;

#define B_   16
#define S_   1024
#define D_   1024
#define H_   16
#define HD_  64
#define SCALE_ 0.125f   // 64^-0.5

struct Bytes32 { uint4 lo, hi; };

// fp32 -> bf16 round-to-nearest-even
static __device__ __forceinline__ u16 f2bf_u16(float f) {
  unsigned u = __builtin_bit_cast(unsigned, f);
  unsigned r = u + 0x7FFFu + ((u >> 16) & 1u);
  return (u16)(r >> 16);
}

// Load a 16-element bf16 fragment from two contiguous 16B chunks.
static __device__ __forceinline__ bf16x16 load_frag(const u16* p0, const u16* p1) {
  Bytes32 t;
  t.lo = *reinterpret_cast<const uint4*>(p0);
  t.hi = *reinterpret_cast<const uint4*>(p1);
  return __builtin_bit_cast(bf16x16, t);
}

static __device__ __forceinline__ f32x8 wmma_bf16(bf16x16 a, bf16x16 b, f32x8 c) {
#if defined(__AMDGCN__)
  return __builtin_amdgcn_wmma_f32_16x16x32_bf16(false, a, false, b, (short)0, c,
                                                 false, false);
#else
  (void)a; (void)b;
  return c;
#endif
}

// ---------------------------------------------------------------------------
// Kernel 0a: convert+transpose 4 weight matrices fp32[K,N] -> bf16[N,K]
// grid = 4 * (1024/32) * (1024/32) = 4096 blocks, 256 threads
// ---------------------------------------------------------------------------
__global__ __launch_bounds__(256)
void k_weight_cvt(const float* __restrict__ w0, const float* __restrict__ w1,
                  const float* __restrict__ w2, const float* __restrict__ w3,
                  u16* __restrict__ dst) {
  __shared__ float tile[32][33];
  int mat = blockIdx.x >> 10;          // 1024 tiles per matrix
  int t   = blockIdx.x & 1023;
  int tk  = t >> 5, tn = t & 31;
  const float* src = (mat == 0) ? w0 : (mat == 1) ? w1 : (mat == 2) ? w2 : w3;
  u16* out = dst + (size_t)mat * D_ * D_;
  int x = threadIdx.x & 31;
  int y = threadIdx.x >> 5;            // 0..7
#pragma unroll
  for (int j = 0; j < 4; ++j) {
    int k = tk * 32 + y + j * 8;
    tile[y + j * 8][x] = src[(size_t)k * D_ + tn * 32 + x];
  }
  __syncthreads();
#pragma unroll
  for (int j = 0; j < 4; ++j) {
    int n = tn * 32 + y + j * 8;
    out[(size_t)n * D_ + tk * 32 + x] = f2bf_u16(tile[x][y + j * 8]);
  }
}

// ---------------------------------------------------------------------------
// Kernel 0b: x fp32 -> bf16, same layout [B*S, D].  4 elems / thread.
// ---------------------------------------------------------------------------
__global__ __launch_bounds__(256)
void k_x_cvt(const float* __restrict__ x, u16* __restrict__ xb) {
  size_t i = ((size_t)blockIdx.x * blockDim.x + threadIdx.x) * 4;
  float4 v = *reinterpret_cast<const float4*>(x + i);
  uint2 p;
  p.x = (unsigned)f2bf_u16(v.x) | ((unsigned)f2bf_u16(v.y) << 16);
  p.y = (unsigned)f2bf_u16(v.z) | ((unsigned)f2bf_u16(v.w) << 16);
  *reinterpret_cast<uint2*>(xb + i) = p;
}

// ---------------------------------------------------------------------------
// Kernel 1: QKV projections. C[16384,1024] = xb @ W + b, per matrix.
// Wave computes a 16x64 strip.  Q,K stored bf16 [B,H,S,HD]; V stored
// transposed [B,H,HD,S] so the AV B-fragment is contiguous.
// waves = 3 * 1024 * 16 = 49152 -> 6144 blocks
// ---------------------------------------------------------------------------
__global__ __launch_bounds__(256)
void k_qkv(const u16* __restrict__ xb, const u16* __restrict__ wt,
           const float* __restrict__ bq, const float* __restrict__ bk,
           const float* __restrict__ bv,
           u16* __restrict__ Qb, u16* __restrict__ Kb, u16* __restrict__ Vt) {
  int wave = blockIdx.x * 8 + (threadIdx.x >> 5);
  int lane = threadIdx.x & 31;
  int r16 = lane & 15, half = lane >> 4;
  int mat = wave >> 14;                // 16384 waves per matrix
  int rem = wave & 16383;
  int mt  = rem >> 4;                  // 0..1023  (M tile)
  int nt  = rem & 15;                  // 0..15    (64-wide N strip)
  const u16* W = wt + (size_t)mat * D_ * D_;
  const float* bias = (mat == 0) ? bq : (mat == 1) ? bk : bv;
  int m0 = mt * 16;
  int arow = m0 + r16;

  f32x8 acc[4];
#pragma unroll
  for (int j = 0; j < 4; ++j) {
    float bb = bias[nt * 64 + j * 16 + r16];
#pragma unroll
    for (int v = 0; v < 8; ++v) acc[j][v] = bb;
  }

  for (int k0 = 0; k0 < D_; k0 += 32) {
    const u16* ap = xb + (size_t)arow * D_ + k0;
    bf16x16 a = load_frag(ap + half * 8, ap + 16 + half * 8);
#pragma unroll
    for (int j = 0; j < 4; ++j) {
      int col = nt * 64 + j * 16 + r16;
      const u16* bp = W + (size_t)col * D_ + k0;
      acc[j] = wmma_bf16(a, load_frag(bp + half * 8, bp + 16 + half * 8), acc[j]);
    }
  }

#pragma unroll
  for (int j = 0; j < 4; ++j) {
    int n = nt * 64 + j * 16 + r16;
    int h = n >> 6, hd = n & 63;
#pragma unroll
    for (int v = 0; v < 8; ++v) {
      int m = m0 + v + 8 * half;
      int b = m >> 10, s = m & 1023;
      u16 val = f2bf_u16(acc[j][v]);
      if (mat == 0)
        Qb[((size_t)(b * H_ + h) * S_ + s) * HD_ + hd] = val;
      else if (mat == 1)
        Kb[((size_t)(b * H_ + h) * S_ + s) * HD_ + hd] = val;
      else
        Vt[((size_t)(b * H_ + h) * HD_ + hd) * S_ + s] = val;
    }
  }
}

// ---------------------------------------------------------------------------
// Kernel 2: fused scores + softmax + attn@V.
// One workgroup (8 waves) per (b,h, 16-query strip):
//   P1: wave w computes 16x128 score slice (cols w*128..+127) via WMMA,
//       fp32 scores -> LDS [16][1024] (64KB).
//   P2: wave w softmaxes rows 2w,2w+1 (fp32, shfl reductions); writes the
//       normalized row ONCE to global attn (the only HBM pass over attn)
//       and a bf16 copy into LDS low 32KB.
//   P3: wave w computes a 16x64 partial of attn@V over K-slice w*128..+127,
//       A-fragments straight from LDS bf16; partials -> LDS high 32KB;
//       cooperative 8-way reduction -> bf16 Ob.
// grid = 256 bh * 64 qtiles = 16384 blocks, 256 threads, 64KB LDS.
// ---------------------------------------------------------------------------
__global__ __launch_bounds__(256)
void k_attn(const u16* __restrict__ Qb, const u16* __restrict__ Kb,
            const u16* __restrict__ Vt, float* __restrict__ attn,
            u16* __restrict__ Ob) {
  __shared__ __align__(16) char smem_raw[65536];
  float* sF = reinterpret_cast<float*>(smem_raw);          // [16][1024] fp32
  u16*   sB = reinterpret_cast<u16*>(smem_raw);            // [16][1024] bf16
  float* sP = reinterpret_cast<float*>(smem_raw + 32768);  // [8][16][64] fp32

  int bh   = blockIdx.x >> 6;
  int qt   = blockIdx.x & 63;
  int w    = threadIdx.x >> 5;
  int lane = threadIdx.x & 31;
  int r16 = lane & 15, half = lane >> 4;
  int q0 = qt * 16;

  const u16* Q = Qb + (size_t)bh * S_ * HD_;
  const u16* K = Kb + (size_t)bh * S_ * HD_;
  const u16* V = Vt + (size_t)bh * HD_ * S_;

  // ---- Phase 1: scores ----
  {
    f32x8 acc[8] = {};
#pragma unroll
    for (int k0 = 0; k0 < HD_; k0 += 32) {
      const u16* ap = Q + (size_t)(q0 + r16) * HD_ + k0;
      bf16x16 a = load_frag(ap + half * 8, ap + 16 + half * 8);
#pragma unroll
      for (int j = 0; j < 8; ++j) {
        int col = w * 128 + j * 16 + r16;          // key index
        const u16* bp = K + (size_t)col * HD_ + k0;
        acc[j] = wmma_bf16(a, load_frag(bp + half * 8, bp + 16 + half * 8), acc[j]);
      }
    }
#pragma unroll
    for (int j = 0; j < 8; ++j) {
      int col = w * 128 + j * 16 + r16;
#pragma unroll
      for (int v = 0; v < 8; ++v)
        sF[(v + 8 * half) * 1024 + col] = acc[j][v] * SCALE_;
    }
  }
  __syncthreads();

  // ---- Phase 2: softmax (rows 2w, 2w+1) ----
  int ra = 2 * w, rb = 2 * w + 1;
  float4 r0[8], r1[8];
  float m0 = -3.0e38f, m1 = -3.0e38f;
#pragma unroll
  for (int i = 0; i < 8; ++i) {
    r0[i] = reinterpret_cast<const float4*>(sF + ra * 1024)[i * 32 + lane];
    r1[i] = reinterpret_cast<const float4*>(sF + rb * 1024)[i * 32 + lane];
    m0 = fmaxf(m0, fmaxf(fmaxf(r0[i].x, r0[i].y), fmaxf(r0[i].z, r0[i].w)));
    m1 = fmaxf(m1, fmaxf(fmaxf(r1[i].x, r1[i].y), fmaxf(r1[i].z, r1[i].w)));
  }
#pragma unroll
  for (int off = 16; off > 0; off >>= 1) {
    m0 = fmaxf(m0, __shfl_xor(m0, off, 32));
    m1 = fmaxf(m1, __shfl_xor(m1, off, 32));
  }
  float s0 = 0.f, s1 = 0.f;
#pragma unroll
  for (int i = 0; i < 8; ++i) {
    r0[i].x = expf(r0[i].x - m0); r0[i].y = expf(r0[i].y - m0);
    r0[i].z = expf(r0[i].z - m0); r0[i].w = expf(r0[i].w - m0);
    r1[i].x = expf(r1[i].x - m1); r1[i].y = expf(r1[i].y - m1);
    r1[i].z = expf(r1[i].z - m1); r1[i].w = expf(r1[i].w - m1);
    s0 += r0[i].x + r0[i].y + r0[i].z + r0[i].w;
    s1 += r1[i].x + r1[i].y + r1[i].z + r1[i].w;
  }
#pragma unroll
  for (int off = 16; off > 0; off >>= 1) {
    s0 += __shfl_xor(s0, off, 32);
    s1 += __shfl_xor(s1, off, 32);
  }
  float inv0 = 1.0f / s0, inv1 = 1.0f / s1;
  __syncthreads();   // fp32 score region dead; low half reusable as bf16

  float* arow0 = attn + ((size_t)bh * S_ + q0 + ra) * S_;
  float* arow1 = attn + ((size_t)bh * S_ + q0 + rb) * S_;
#pragma unroll
  for (int i = 0; i < 8; ++i) {
    r0[i].x *= inv0; r0[i].y *= inv0; r0[i].z *= inv0; r0[i].w *= inv0;
    r1[i].x *= inv1; r1[i].y *= inv1; r1[i].z *= inv1; r1[i].w *= inv1;
    reinterpret_cast<float4*>(arow0)[i * 32 + lane] = r0[i];
    reinterpret_cast<float4*>(arow1)[i * 32 + lane] = r1[i];
    uint2 p0, p1;
    p0.x = (unsigned)f2bf_u16(r0[i].x) | ((unsigned)f2bf_u16(r0[i].y) << 16);
    p0.y = (unsigned)f2bf_u16(r0[i].z) | ((unsigned)f2bf_u16(r0[i].w) << 16);
    p1.x = (unsigned)f2bf_u16(r1[i].x) | ((unsigned)f2bf_u16(r1[i].y) << 16);
    p1.y = (unsigned)f2bf_u16(r1[i].z) | ((unsigned)f2bf_u16(r1[i].w) << 16);
    *reinterpret_cast<uint2*>(sB + ra * 1024 + (i * 32 + lane) * 4) = p0;
    *reinterpret_cast<uint2*>(sB + rb * 1024 + (i * 32 + lane) * 4) = p1;
  }
  __syncthreads();

  // ---- Phase 3: attn @ V, wave w takes K-slice [w*128, w*128+128) ----
  {
    f32x8 acc[4] = {};
#pragma unroll
    for (int kk = 0; kk < 4; ++kk) {
      int k0 = w * 128 + kk * 32;
      const u16* ap = sB + r16 * 1024 + k0;
      bf16x16 a = load_frag(ap + half * 8, ap + 16 + half * 8);
#pragma unroll
      for (int j = 0; j < 4; ++j) {
        int col = j * 16 + r16;                    // hd
        const u16* bp = V + (size_t)col * S_ + k0;
        acc[j] = wmma_bf16(a, load_frag(bp + half * 8, bp + 16 + half * 8), acc[j]);
      }
    }
#pragma unroll
    for (int j = 0; j < 4; ++j)
#pragma unroll
      for (int v = 0; v < 8; ++v)
        sP[(w * 16 + v + 8 * half) * 64 + j * 16 + r16] = acc[j][v];
  }
  __syncthreads();

  // ---- cooperative 8-way partial reduction -> Ob bf16 ----
  {
    int o = threadIdx.x * 4;                       // 1024 outputs / 256 threads
    float4 s4 = {0.f, 0.f, 0.f, 0.f};
#pragma unroll
    for (int p = 0; p < 8; ++p) {
      float4 t = *reinterpret_cast<const float4*>(sP + p * 1024 + o);
      s4.x += t.x; s4.y += t.y; s4.z += t.z; s4.w += t.w;
    }
    int row = o >> 6, hd = o & 63;
    uint2 pk;
    pk.x = (unsigned)f2bf_u16(s4.x) | ((unsigned)f2bf_u16(s4.y) << 16);
    pk.y = (unsigned)f2bf_u16(s4.z) | ((unsigned)f2bf_u16(s4.w) << 16);
    *reinterpret_cast<uint2*>(Ob + ((size_t)bh * S_ + q0 + row) * HD_ + hd) = pk;
  }
}

// ---------------------------------------------------------------------------
// Kernel 3: out = concat_heads(out_heads) @ Wo + bo   (fp32 result)
// waves = 1024 * 16 = 16384 -> 2048 blocks
// ---------------------------------------------------------------------------
__global__ __launch_bounds__(256)
void k_proj(const u16* __restrict__ Ob, const u16* __restrict__ Wto,
            const float* __restrict__ bo, float* __restrict__ out) {
  int wave = blockIdx.x * 8 + (threadIdx.x >> 5);
  int lane = threadIdx.x & 31;
  int r16 = lane & 15, half = lane >> 4;
  int mt = wave >> 4;                  // 0..1023
  int nt = wave & 15;                  // 0..15
  int m0 = mt * 16;
  int m = m0 + r16;
  int b = m >> 10, s = m & 1023;

  f32x8 acc[4];
#pragma unroll
  for (int j = 0; j < 4; ++j) {
    float bb = bo[nt * 64 + j * 16 + r16];
#pragma unroll
    for (int v = 0; v < 8; ++v) acc[j][v] = bb;
  }

  for (int k0 = 0; k0 < D_; k0 += 32) {
    int ka = k0 + half * 8;
    int kb = k0 + 16 + half * 8;
    const u16* p0 = Ob + ((size_t)(b * H_ + (ka >> 6)) * S_ + s) * HD_ + (ka & 63);
    const u16* p1 = Ob + ((size_t)(b * H_ + (kb >> 6)) * S_ + s) * HD_ + (kb & 63);
    bf16x16 a = load_frag(p0, p1);
#pragma unroll
    for (int j = 0; j < 4; ++j) {
      int col = nt * 64 + j * 16 + r16;
      const u16* bp = Wto + (size_t)col * D_ + k0;
      acc[j] = wmma_bf16(a, load_frag(bp + half * 8, bp + 16 + half * 8), acc[j]);
    }
  }
#pragma unroll
  for (int j = 0; j < 4; ++j) {
    int n = nt * 64 + j * 16 + r16;
#pragma unroll
    for (int v = 0; v < 8; ++v) {
      int mm = m0 + v + 8 * half;
      out[(size_t)mm * D_ + n] = acc[j][v];
    }
  }
}

// ---------------------------------------------------------------------------
extern "C" void kernel_launch(void* const* d_in, const int* in_sizes, int n_in,
                              void* d_out, int out_size, void* d_ws, size_t ws_size,
                              hipStream_t stream) {
  (void)in_sizes; (void)n_in; (void)out_size; (void)ws_size;
  const float* x  = (const float*)d_in[0];
  const float* Wq = (const float*)d_in[1];
  const float* bq = (const float*)d_in[2];
  const float* Wk = (const float*)d_in[3];
  const float* bk = (const float*)d_in[4];
  const float* Wv = (const float*)d_in[5];
  const float* bv = (const float*)d_in[6];
  const float* Wo = (const float*)d_in[7];
  const float* bo = (const float*)d_in[8];

  float* out  = (float*)d_out;
  float* attn = out + (size_t)B_ * S_ * D_;   // [B,H,S,S] fp32

  char* ws = (char*)d_ws;
  const size_t WBYTES = (size_t)D_ * D_ * sizeof(u16);        // 2 MB
  const size_t XBYTES = (size_t)B_ * S_ * D_ * sizeof(u16);   // 32 MB
  u16* wt  = (u16*)(ws);                             // Wq^T,Wk^T,Wv^T,Wo^T bf16
  u16* wto = (u16*)(ws + 3 * WBYTES);
  u16* xb  = (u16*)(ws + 4 * WBYTES);
  u16* Qb  = (u16*)(ws + 4 * WBYTES + 1 * XBYTES);   // [B,H,S,HD]
  u16* Kb  = (u16*)(ws + 4 * WBYTES + 2 * XBYTES);   // [B,H,S,HD]
  u16* Vt  = (u16*)(ws + 4 * WBYTES + 3 * XBYTES);   // [B,H,HD,S]
  u16* Ob  = (u16*)(ws + 4 * WBYTES + 4 * XBYTES);   // [B,H,S,HD]

  k_weight_cvt<<<4096, 256, 0, stream>>>(Wq, Wk, Wv, Wo, wt);
  k_x_cvt<<<(B_ * S_ * D_) / (256 * 4), 256, 0, stream>>>(x, xb);
  k_qkv<<<6144, 256, 0, stream>>>(xb, wt, bq, bk, bv, Qb, Kb, Vt);
  k_attn<<<16384, 256, 0, stream>>>(Qb, Kb, Vt, attn, Ob);
  k_proj<<<2048, 256, 0, stream>>>(Ob, wto, bo, out);
}
